// model_84413287235748
// MI455X (gfx1250) — matrix operand order repaired and checked by reading, not compiled, not run on hardware
//
#include <hip/hip_runtime.h>
#include <cstdint>
#include <cstddef>

typedef __attribute__((ext_vector_type(16))) __bf16 v16bf;
typedef __attribute__((ext_vector_type(8)))  float  v8f;

union BfBlk { unsigned int u[8]; v16bf v; };

__device__ __forceinline__ unsigned int packbf_hw(float a, float b) {  // lo=bf16(a), hi=bf16(b)
    unsigned int r;
    asm("v_cvt_pk_bf16_f32 %0, %1, %2" : "=v"(r) : "v"(a), "v"(b));
    return r;
}
// 4 transcendentals per asm block: consecutive v_tanh ops are each other's
// independent-op hazard filler; single trailing v_nop covers the last one.
__device__ __forceinline__ void tanh4(float& r0, float& r1, float& r2, float& r3,
                                      float a0, float a1, float a2, float a3) {
    asm volatile("v_tanh_f32 %0, %4\n\t"
                 "v_tanh_f32 %1, %5\n\t"
                 "v_tanh_f32 %2, %6\n\t"
                 "v_tanh_f32 %3, %7\n\t"
                 "v_nop"
                 : "=&v"(r0), "=&v"(r1), "=&v"(r2), "=&v"(r3)
                 : "v"(a0), "v"(a1), "v"(a2), "v"(a3));
}
__device__ __forceinline__ v8f wmma_bf16(const unsigned int* au, const unsigned int* bu, v8f c) {
    BfBlk a, b;
#pragma unroll
    for (int i = 0; i < 8; ++i) { a.u[i] = au[i]; b.u[i] = bu[i]; }
    return __builtin_amdgcn_wmma_f32_16x16x32_bf16(false, a.v, false, b.v, (short)0, c, false, false);
}

#define NN 65536
#define HH 128
#define MM 32

// LDS weight image: 20 tiles, each tile = per-lane 8 dwords (bf16 pairs) in exact
// WMMA A-operand register layout. Tiles 0..7: A_R (R-projection, used once),
// 8..15: layer-1 dynamic-feature tiles, 16..19: layer-2 (permuted W2^T) chunks.
#define T_AR  0
#define T_A1  8
#define T_A2  16

__global__ __launch_bounds__(256)
void sde_mlp_kernel(
    const float* __restrict__ R,  const float* __restrict__ X0,
    const float* __restrict__ V0, const float* __restrict__ noise,
    const float* __restrict__ W1, const float* __restrict__ b1,
    const float* __restrict__ W2, const float* __restrict__ b2,
    float* __restrict__ out)
{
    const float dt   = 1.0f / 32.0f;
    const float sqdt = 0.17677669529663687f;   // sqrt(1/32)

    __shared__ uint4 wtile[20][64];            // [tile][lane*2 + j], 20 KB

    const int lane = threadIdx.x & 31;
    const int wave = blockIdx.x * (blockDim.x >> 5) + (threadIdx.x >> 5);
    const int half = lane >> 4;                // which 16-lane half of the wave
    const int mrow = lane & 15;                // A-matrix row / tile-local particle id
    const int p    = wave * 16 + mrow;         // lanes 16-31 mirror lanes 0-15's particle
    const bool lo  = (lane < 16);

    // ------------- wave 0 stages all WMMA A-tiles into LDS (once per block) -------------
    if ((threadIdx.x >> 5) == 0) {
        unsigned int u[8];

        // A_R tiles: element (m,k) = W1[k][16t+m], k = 0..31 (R-feature rows of W1)
#pragma unroll
        for (int t = 0; t < 8; ++t) {
            const int h = 16 * t + mrow;
#pragma unroll
            for (int v = 0; v < 8; ++v) {
                const int k0 = (v < 4) ? (2 * v + 8 * half) : (16 + 2 * (v - 4) + 8 * half);
                u[v] = packbf_hw(W1[k0 * HH + h], W1[(k0 + 1) * HH + h]);
            }
            wtile[T_AR + t][lane * 2]     = make_uint4(u[0], u[1], u[2], u[3]);
            wtile[T_AR + t][lane * 2 + 1] = make_uint4(u[4], u[5], u[6], u[7]);
        }

        // Layer-1 dynamic tiles: (m,k) = W1[33+k][h] for k<8, W1[32][h] for k==8, else 0
#pragma unroll
        for (int t = 0; t < 8; ++t) {
            const int h = 16 * t + mrow;
#pragma unroll
            for (int v = 0; v < 8; ++v) {
                const int k0 = (v < 4) ? (2 * v + 8 * half) : (16 + 2 * (v - 4) + 8 * half);
                const int k1 = k0 + 1;
                const int r0 = (k0 < 8) ? (33 + k0) : 32;
                const int r1 = (k1 < 8) ? (33 + k1) : 32;
                float f0 = W1[r0 * HH + h]; if (k0 > 8) f0 = 0.0f;
                float f1 = W1[r1 * HH + h]; if (k1 > 8) f1 = 0.0f;
                u[v] = packbf_hw(f0, f1);
            }
            wtile[T_A1 + t][lane * 2]     = make_uint4(u[0], u[1], u[2], u[3]);
            wtile[T_A1 + t][lane * 2 + 1] = make_uint4(u[4], u[5], u[6], u[7]);
        }

        // Layer-2 chunks (W2^T) with the hidden-dim permutation induced by feeding
        // layer-1 C-tiles straight into the B operand (no cross-lane shuffles):
        //   hw-K k: sub=k>>3, r=k&7 -> tile = 2c+(sub>>1), mrow' = (sub&1)*8+r
#pragma unroll
        for (int c = 0; c < 4; ++c) {
#pragma unroll
            for (int v = 0; v < 8; ++v) {
                const int kb = (v < 4) ? (2 * v + 8 * half) : (16 + 2 * (v - 4) + 8 * half);
                float fv[2];
#pragma unroll
                for (int kk = 0; kk < 2; ++kk) {
                    const int k   = kb + kk;
                    const int sub = k >> 3, r = k & 7;
                    const int hid = 16 * (2 * c + (sub >> 1)) + ((sub & 1) << 3) + r;
                    fv[kk] = (mrow < 8) ? W2[hid * 8 + mrow] : 0.0f;  // pad rows 8..15
                }
                u[v] = packbf_hw(fv[0], fv[1]);
            }
            wtile[T_A2 + c][lane * 2]     = make_uint4(u[0], u[1], u[2], u[3]);
            wtile[T_A2 + c][lane * 2 + 1] = make_uint4(u[4], u[5], u[6], u[7]);
        }
    }
    __syncthreads();

    // ------------- one-time R projection: racc[t] = (R @ W1_R)^T + b1 -------------
    unsigned int br[8];
    {
        const float* Rp = R + (size_t)p * 32 + half * 8;    // lanes 0-15: K 0-7,16-23; 16-31: K 8-15,24-31
        float rr[16];
#pragma unroll
        for (int j = 0; j < 4; ++j) {
            float4 q = *reinterpret_cast<const float4*>(Rp + 4 * j + (j >> 1) * 8);
            rr[4 * j + 0] = q.x; rr[4 * j + 1] = q.y; rr[4 * j + 2] = q.z; rr[4 * j + 3] = q.w;
        }
#pragma unroll
        for (int v = 0; v < 8; ++v) br[v] = packbf_hw(rr[2 * v], rr[2 * v + 1]);
    }

    v8f racc[8];
#pragma unroll
    for (int t = 0; t < 8; ++t) {
        unsigned int ar[8];
        uint4 q0 = wtile[T_AR + t][lane * 2];
        uint4 q1 = wtile[T_AR + t][lane * 2 + 1];
        ar[0] = q0.x; ar[1] = q0.y; ar[2] = q0.z; ar[3] = q0.w;
        ar[4] = q1.x; ar[5] = q1.y; ar[6] = q1.z; ar[7] = q1.w;

        const float* bp = b1 + 16 * t + 8 * half;           // lane-uniform within a half
        float4 c0 = *reinterpret_cast<const float4*>(bp);
        float4 c1 = *reinterpret_cast<const float4*>(bp + 4);
        v8f cb = { c0.x, c0.y, c0.z, c0.w, c1.x, c1.y, c1.z, c1.w };
        racc[t] = wmma_bf16(ar, br, cb);
    }

    // ---------------- per-particle state ----------------
    float b2l[8];
    {
        float4 q0 = *reinterpret_cast<const float4*>(b2);
        float4 q1 = *reinterpret_cast<const float4*>(b2 + 4);
        b2l[0] = q0.x; b2l[1] = q0.y; b2l[2] = q0.z; b2l[3] = q0.w;
        b2l[4] = q1.x; b2l[5] = q1.y; b2l[6] = q1.z; b2l[7] = q1.w;
    }
    float X[8];
    {
        float4 x0 = *reinterpret_cast<const float4*>(X0 + (size_t)p * 8);
        float4 x1 = *reinterpret_cast<const float4*>(X0 + (size_t)p * 8 + 4);
        X[0] = x0.x; X[1] = x0.y; X[2] = x0.z; X[3] = x0.w;
        X[4] = x1.x; X[5] = x1.y; X[6] = x1.z; X[7] = x1.w;
    }
    float V = V0[p];

    // ---------------- time stepping ----------------
    for (int i = 0; i < MM; ++i) {
        const float s = (float)i * dt;

        // B operand for layer 1: features x particles; K 0-7 = X, K 8 = s, rest 0.
        unsigned int bx[8];
#pragma unroll
        for (int v = 0; v < 4; ++v) bx[v] = lo ? packbf_hw(X[2 * v], X[2 * v + 1]) : (v == 0 ? packbf_hw(s, 0.0f) : 0u);
        bx[4] = 0u;
        bx[5] = 0u; bx[6] = 0u; bx[7] = 0u;

        v8f zacc = { 0.f, 0.f, 0.f, 0.f, 0.f, 0.f, 0.f, 0.f };
#pragma unroll
        for (int c = 0; c < 4; ++c) {
            unsigned int aa[8], ab[8], a2c[8];
            {
                uint4 q0 = wtile[T_A1 + 2 * c][lane * 2];
                uint4 q1 = wtile[T_A1 + 2 * c][lane * 2 + 1];
                aa[0] = q0.x; aa[1] = q0.y; aa[2] = q0.z; aa[3] = q0.w;
                aa[4] = q1.x; aa[5] = q1.y; aa[6] = q1.z; aa[7] = q1.w;
            }
            {
                uint4 q0 = wtile[T_A1 + 2 * c + 1][lane * 2];
                uint4 q1 = wtile[T_A1 + 2 * c + 1][lane * 2 + 1];
                ab[0] = q0.x; ab[1] = q0.y; ab[2] = q0.z; ab[3] = q0.w;
                ab[4] = q1.x; ab[5] = q1.y; ab[6] = q1.z; ab[7] = q1.w;
            }
            v8f va = wmma_bf16(aa, bx, racc[2 * c]);
            v8f vb = wmma_bf16(ab, bx, racc[2 * c + 1]);

            unsigned int bh[8];
#pragma unroll
            for (int j = 0; j < 4; ++j) {
                float t0, t1, t2, t3;
                tanh4(t0, t1, t2, t3, va[2 * j], va[2 * j + 1], vb[2 * j], vb[2 * j + 1]);
                bh[j]     = packbf_hw(t0, t1);
                bh[4 + j] = packbf_hw(t2, t3);
            }
            {
                uint4 q0 = wtile[T_A2 + c][lane * 2];
                uint4 q1 = wtile[T_A2 + c][lane * 2 + 1];
                a2c[0] = q0.x; a2c[1] = q0.y; a2c[2] = q0.z; a2c[3] = q0.w;
                a2c[4] = q1.x; a2c[5] = q1.y; a2c[6] = q1.z; a2c[7] = q1.w;
            }
            zacc = wmma_bf16(a2c, bh, zacc);   // Z^T accumulation over K=128
        }

        // elementwise updates (lanes 0-15 carry real data; 16-31 mirror harmlessly)
        const float* ep = noise + ((size_t)i * NN + p) * 8;
        float4 e0 = *reinterpret_cast<const float4*>(ep);
        float4 e1 = *reinterpret_cast<const float4*>(ep + 4);
        float Wn[8] = { sqdt * e0.x, sqdt * e0.y, sqdt * e0.z, sqdt * e0.w,
                        sqdt * e1.x, sqdt * e1.y, sqdt * e1.z, sqdt * e1.w };
        if (i + 1 < MM)
            __builtin_prefetch(noise + ((size_t)(i + 1) * NN + p) * 8, 0, 1);

        float zz = 0.0f, zw = 0.0f;
#pragma unroll
        for (int d = 0; d < 8; ++d) {
            const float z = zacc[d] + b2l[d];
            zz = fmaf(z, z, zz);
            zw = fmaf(z, Wn[d], zw);
            X[d] = fmaf(-dt, X[d] + z, X[d]) + Wn[d];   // X += dt*(-X - Z) + W
        }
        V = fmaf(dt, -0.5f * zz, V) + zw;               // V += dt*(-0.5*|Z|^2) + Z.W
    }

    if (lo) {
        float* xo = out + (size_t)p * 8;
        float4 o0 = { X[0], X[1], X[2], X[3] };
        float4 o1 = { X[4], X[5], X[6], X[7] };
        *reinterpret_cast<float4*>(xo)     = o0;
        *reinterpret_cast<float4*>(xo + 4) = o1;
        out[(size_t)NN * 8 + p] = V;
    }
}

extern "C" void kernel_launch(void* const* d_in, const int* in_sizes, int n_in,
                              void* d_out, int out_size, void* d_ws, size_t ws_size,
                              hipStream_t stream) {
    const float* R     = (const float*)d_in[0];
    const float* X0    = (const float*)d_in[1];
    const float* V0    = (const float*)d_in[2];
    const float* noise = (const float*)d_in[3];
    const float* W1    = (const float*)d_in[4];
    const float* b1    = (const float*)d_in[5];
    const float* W2    = (const float*)d_in[6];
    const float* b2    = (const float*)d_in[7];
    float* out = (float*)d_out;
    (void)in_sizes; (void)n_in; (void)out_size; (void)d_ws; (void)ws_size;

    dim3 block(256);               // 8 waves -> 128 particles per block
    dim3 grid(NN / 128);           // 512 blocks
    hipLaunchKernelGGL(sde_mlp_kernel, grid, block, 0, stream,
                       R, X0, V0, noise, W1, b1, W2, b2, out);
}
